// GATv2Model_30116310679911
// MI455X (gfx1250) — compile-verified, matrix operand order checked by reading, not run on hardware
//
#include <hip/hip_runtime.h>

#define DEV __device__ __forceinline__

typedef __attribute__((ext_vector_type(16))) __bf16 v16bf;
typedef __attribute__((ext_vector_type(8)))  __bf16 v8bf;
typedef __attribute__((ext_vector_type(4)))  __bf16 v4bf;
typedef __attribute__((ext_vector_type(8)))  float  v8f;
typedef __attribute__((ext_vector_type(2)))  float  v2f;
typedef __attribute__((ext_vector_type(4)))  unsigned int u32x4;
typedef __attribute__((ext_vector_type(8)))  unsigned int u32x8;

constexpr int N_NODES   = 100000;
constexpr int IN_C      = 1433;
constexpr int DIM       = 32;
constexpr int HEADS     = 8;
constexpr int OUT_C     = 4;
constexpr int HC        = HEADS * OUT_C;   // 32
constexpr int NUM_CLASS = 7;
constexpr float NEG_SLOPE = 0.2f;

constexpr int NSLICE   = (IN_C + 31) / 32;   // 45 k-slices of 32
constexpr int ROWS_BLK = 128;                // 8 waves x 16 rows
constexpr int XROW_PAD = 44;                 // 32 data + 12 pad floats (bank-conflict-free)

// ---------------------------------------------------------------------------
// monotonic float <-> uint encoding for atomicMax-based segment max
// ---------------------------------------------------------------------------
DEV unsigned enc_f32(float f) {
    unsigned u = __float_as_uint(f);
    return (u & 0x80000000u) ? ~u : (u | 0x80000000u);
}
DEV float dec_f32(unsigned u) {
    return (u & 0x80000000u) ? __uint_as_float(u & 0x7FFFFFFFu)
                             : __uint_as_float(~u);
}
// enc(-inf) = 0x007FFFFF

// ---------------------------------------------------------------------------
// Issue one TDM 2D tile load: x[rowBase .. rowBase+127, k0 .. k0+31] -> LDS,
// rows padded 32->44 floats in LDS; hardware zero-fills K-tail and row-tail.
// ---------------------------------------------------------------------------
DEV void tdm_load_x_tile(const float* x, int rowBase, int k0, unsigned lds_byte_addr)
{
    unsigned long long ga =
        (unsigned long long)(uintptr_t)x +
        ((unsigned long long)rowBase * IN_C + (unsigned)k0) * 4ull;

    unsigned td0 = (unsigned)(IN_C - k0);        // remaining elements in dim0
    unsigned td1 = (unsigned)(N_NODES - rowBase);// remaining rows in dim1

    u32x4 g0;
    g0[0] = 1u;                                   // count=1, user mode, no gather
    g0[1] = lds_byte_addr;                        // D#.lds_addr
    g0[2] = (unsigned)ga;                         // global_addr[31:0]
    g0[3] = (unsigned)((ga >> 32) & 0x1FFFFFFu) | (2u << 30);  // addr[56:32] | type=2

    u32x8 g1;
    g1[0] = (2u << 16)      // data_size = 4 bytes
          | (1u << 20)      // pad_enable
          | (4u << 22)      // pad_interval code 4 -> every 32 DWORDs (one 32-float row)
          | (11u << 25);    // pad_amount  code 11 -> 12 DWORDs (row 32 -> 44 floats)
    g1[1] = (td0 & 0xFFFFu) << 16;                        // tensor_dim0[15:0]
    g1[2] = (td0 >> 16) | ((td1 & 0xFFFFu) << 16);        // tensor_dim0[31:16] | tensor_dim1[15:0]
    g1[3] = (td1 >> 16) | (32u << 16);                    // tensor_dim1[31:16] | tile_dim0=32
    g1[4] = (unsigned)ROWS_BLK;                           // tile_dim1=128, tile_dim2=0
    g1[5] = (unsigned)IN_C;                               // tensor_dim0_stride = 1433 elements
    g1[6] = 0u;
    g1[7] = 0u;

    asm volatile("tensor_load_to_lds %0, %1" :: "s"(g0), "s"(g1) : "memory");
}

// ---------------------------------------------------------------------------
// Kernel 1: h = x @ w_proj + b_proj   (N x 1433) @ (1433 x 32)
// TDM double-buffered A tiles in LDS, swizzled bf16 B slice in LDS,
// bf16 WMMA with f32 accumulate. 8 waves/block, 16 rows/wave.
// ---------------------------------------------------------------------------
__global__ void __launch_bounds__(256) proj_gemm_kernel(
    const float* __restrict__ x, const float* __restrict__ w,
    const float* __restrict__ bias, float* __restrict__ h)
{
    // x tiles: 2 buffers x 128 rows x 44 floats = 45056 B
    __shared__ __align__(16) float  sX[2][ROWS_BLK * XROW_PAD];
    // w slice, bf16, 16B-chunk swizzle: chunk(kb,n) = kb*32+n, kb=k/8 (0..3)
    __shared__ __align__(16) __bf16 sW[4 * 32 * 8];               // 2048 B

    const int tid  = threadIdx.x;
    const int lane = tid & 31;
    const int wave = tid >> 5;
    const int ln   = lane & 15;
    const int hi   = lane >> 4;
    const int rowBase = blockIdx.x * ROWS_BLK;

    const unsigned ldsX0 = (unsigned)(uintptr_t)(void*)&sX[0][0];
    const unsigned ldsX1 = (unsigned)(uintptr_t)(void*)&sX[1][0];

    // prologue: prefetch slice 0
    if (wave == 0) tdm_load_x_tile(x, rowBase, 0, ldsX0);

    v8f acc0 = {};
    v8f acc1 = {};

    const v8bf* sWv = (const v8bf*)sW;
    const float* myRow0 = &sX[0][(wave * 16 + ln) * XROW_PAD];
    const float* myRow1 = &sX[1][(wave * 16 + ln) * XROW_PAD];

    for (int s = 0; s < NSLICE; ++s) {
        const int k0 = s * 32;

        // issue prefetch of slice s+1 as early as possible
        if (wave == 0 && s + 1 < NSLICE)
            tdm_load_x_tile(x, rowBase, k0 + 32, (s & 1) ? ldsX0 : ldsX1);

        // all threads: stage w_proj slice into swizzled bf16 LDS.
        // thread t: column n = t&31, k-quad k4 = t>>5 (coalesced across lanes).
        // Block-uniform guard: only the final slice takes the masked path.
        {
            const int n  = tid & 31;
            const int k4 = tid >> 5;            // 0..7
            const int k  = k0 + 4 * k4;
            float f0, f1, f2, f3;
            if (k0 + 32 <= IN_C) {              // uniform: full slice, no guards
                f0 = w[(k    ) * DIM + n];
                f1 = w[(k + 1) * DIM + n];
                f2 = w[(k + 2) * DIM + n];
                f3 = w[(k + 3) * DIM + n];
            } else {                            // last slice only
                f0 = (k     < IN_C) ? w[(k    ) * DIM + n] : 0.0f;
                f1 = (k + 1 < IN_C) ? w[(k + 1) * DIM + n] : 0.0f;
                f2 = (k + 2 < IN_C) ? w[(k + 2) * DIM + n] : 0.0f;
                f3 = (k + 3 < IN_C) ? w[(k + 3) * DIM + n] : 0.0f;
            }
            v4bf wv;
            wv[0] = (__bf16)f0; wv[1] = (__bf16)f1;
            wv[2] = (__bf16)f2; wv[3] = (__bf16)f3;
            const int kb = k4 >> 1, kin4 = (k4 & 1) * 4;
            *(v4bf*)&sW[(kb * 32 + n) * 8 + kin4] = wv;
        }

        // wave0: wait for current tile (prefetch for s+1 may stay outstanding)
        if (wave == 0) {
            if (s + 1 < NSLICE) __builtin_amdgcn_s_wait_tensorcnt(1);
            else                __builtin_amdgcn_s_wait_tensorcnt(0);
        }

        __syncthreads();   // x tile s + w slice s visible to every wave

        // A fragment: 4x ds_load_b128 (conflict-free via 44-float row pad), cvt to bf16
        const float* xb = (s & 1) ? myRow1 : myRow0;
        float4 a0 = *(const float4*)(xb + 8 * hi);
        float4 a1 = *(const float4*)(xb + 8 * hi + 4);
        float4 a2 = *(const float4*)(xb + 16 + 8 * hi);
        float4 a3 = *(const float4*)(xb + 16 + 8 * hi + 4);
        v16bf a;
        a[0]  = (__bf16)a0.x; a[1]  = (__bf16)a0.y; a[2]  = (__bf16)a0.z; a[3]  = (__bf16)a0.w;
        a[4]  = (__bf16)a1.x; a[5]  = (__bf16)a1.y; a[6]  = (__bf16)a1.z; a[7]  = (__bf16)a1.w;
        a[8]  = (__bf16)a2.x; a[9]  = (__bf16)a2.y; a[10] = (__bf16)a2.z; a[11] = (__bf16)a2.w;
        a[12] = (__bf16)a3.x; a[13] = (__bf16)a3.y; a[14] = (__bf16)a3.z; a[15] = (__bf16)a3.w;

        // B fragments: 4x ds_load_b128 of 16B chunks
        v8bf b0lo = sWv[(hi    ) * 32 + ln];
        v8bf b0hi = sWv[(2 + hi) * 32 + ln];
        v8bf b1lo = sWv[(hi    ) * 32 + 16 + ln];
        v8bf b1hi = sWv[(2 + hi) * 32 + 16 + ln];
        v16bf b0 = __builtin_shufflevector(b0lo, b0hi, 0,1,2,3,4,5,6,7,8,9,10,11,12,13,14,15);
        v16bf b1 = __builtin_shufflevector(b1lo, b1hi, 0,1,2,3,4,5,6,7,8,9,10,11,12,13,14,15);

        acc0 = __builtin_amdgcn_wmma_f32_16x16x32_bf16(false, a, false, b0,
                                                       (short)0, acc0, false, false);
        acc1 = __builtin_amdgcn_wmma_f32_16x16x32_bf16(false, a, false, b1,
                                                       (short)0, acc1, false, false);

        __syncthreads();   // done reading sW + sX[s&1] before they are overwritten
    }

    // D layout: VGPR j -> row j + 8*hi, col = ln
    const int waveRow = rowBase + wave * 16;
#pragma unroll
    for (int j = 0; j < 8; ++j) {
        int r = waveRow + j + 8 * hi;
        if (r < N_NODES) {
            h[r * DIM + ln]      = acc0[j] + bias[ln];
            h[r * DIM + 16 + ln] = acc1[j] + bias[16 + ln];
        }
    }
}

// ---------------------------------------------------------------------------
// Kernel 2: xl = h@wl+bl, xr = h@wr+br   (N x 32) @ (32 x 32), exact f32 WMMA
// ---------------------------------------------------------------------------
__global__ void __launch_bounds__(256) xlxr_kernel(
    const float* __restrict__ h,
    const float* __restrict__ wl, const float* __restrict__ bl,
    const float* __restrict__ wr, const float* __restrict__ br,
    float* __restrict__ xl, float* __restrict__ xr)
{
    const int tid = threadIdx.x, lane = tid & 31, wave = tid >> 5;
    const int ln = lane & 15, hi = lane >> 4;
    int row = blockIdx.x * 128 + wave * 16 + ln;
    int rc  = row < N_NODES ? row : (N_NODES - 1);

    float hv[32];
    const float4* hp = (const float4*)(h + (long)rc * DIM);
#pragma unroll
    for (int q = 0; q < 8; ++q) {
        float4 t = hp[q];
        hv[4*q] = t.x; hv[4*q+1] = t.y; hv[4*q+2] = t.z; hv[4*q+3] = t.w;
    }

    v8f aL0 = {}, aL1 = {}, aR0 = {}, aR1 = {};
#pragma unroll
    for (int s = 0; s < 8; ++s) {
        int k = 4 * s + 2 * hi;                 // f32 A: lanes<16 hold K=k0,k0+1
        v2f a;  a[0] = hv[k]; a[1] = hv[k + 1];
        v2f bL0, bL1, bR0, bR1;
        bL0[0] = wl[k * HC + ln];        bL0[1] = wl[(k + 1) * HC + ln];
        bL1[0] = wl[k * HC + 16 + ln];   bL1[1] = wl[(k + 1) * HC + 16 + ln];
        bR0[0] = wr[k * HC + ln];        bR0[1] = wr[(k + 1) * HC + ln];
        bR1[0] = wr[k * HC + 16 + ln];   bR1[1] = wr[(k + 1) * HC + 16 + ln];
        aL0 = __builtin_amdgcn_wmma_f32_16x16x4_f32(false, a, false, bL0, (short)0, aL0, false, false);
        aL1 = __builtin_amdgcn_wmma_f32_16x16x4_f32(false, a, false, bL1, (short)0, aL1, false, false);
        aR0 = __builtin_amdgcn_wmma_f32_16x16x4_f32(false, a, false, bR0, (short)0, aR0, false, false);
        aR1 = __builtin_amdgcn_wmma_f32_16x16x4_f32(false, a, false, bR1, (short)0, aR1, false, false);
    }

    const int baseRow = blockIdx.x * 128 + wave * 16;
#pragma unroll
    for (int j = 0; j < 8; ++j) {
        int r = baseRow + j + 8 * hi;
        if (r < N_NODES) {
            xl[r * HC + ln]      = aL0[j] + bl[ln];
            xl[r * HC + 16 + ln] = aL1[j] + bl[16 + ln];
            xr[r * HC + ln]      = aR0[j] + br[ln];
            xr[r * HC + 16 + ln] = aR1[j] + br[16 + ln];
        }
    }
}

// ---------------------------------------------------------------------------
// Edge-phase helpers: recompute the 8 per-head logits (xl/xr are L2-resident)
// ---------------------------------------------------------------------------
DEV void compute_logits(const float* __restrict__ xl, const float* __restrict__ xr,
                        int s, int d, const float* __restrict__ att,
                        float* xls, float* lg)
{
    const float4* ap = (const float4*)(xl + (long)s * HC);
    const float4* bp = (const float4*)(xr + (long)d * HC);
#pragma unroll
    for (int h = 0; h < HEADS; ++h) {
        float4 av = ap[h], bv = bp[h];
        float e0 = av.x + bv.x, e1 = av.y + bv.y;
        float e2 = av.z + bv.z, e3 = av.w + bv.w;
        e0 = e0 > 0.0f ? e0 : NEG_SLOPE * e0;
        e1 = e1 > 0.0f ? e1 : NEG_SLOPE * e1;
        e2 = e2 > 0.0f ? e2 : NEG_SLOPE * e2;
        e3 = e3 > 0.0f ? e3 : NEG_SLOPE * e3;
        lg[h] = e0 * att[h*4+0] + e1 * att[h*4+1] + e2 * att[h*4+2] + e3 * att[h*4+3];
        xls[4*h+0] = av.x; xls[4*h+1] = av.y; xls[4*h+2] = av.z; xls[4*h+3] = av.w;
    }
}

__global__ void __launch_bounds__(256) edge_max_kernel(
    const int* __restrict__ src, const int* __restrict__ dst,
    const float* __restrict__ xl, const float* __restrict__ xr,
    const float* __restrict__ att, unsigned* __restrict__ m_enc, int E)
{
    int e = blockIdx.x * 256 + threadIdx.x;
    if (e >= E) return;
    int s = src[e], d = dst[e];
    float xls[32], lg[8];
    compute_logits(xl, xr, s, d, att, xls, lg);
#pragma unroll
    for (int h = 0; h < HEADS; ++h)
        atomicMax(&m_enc[d * HEADS + h], enc_f32(lg[h]));
}

__global__ void __launch_bounds__(256) edge_sum_kernel(
    const int* __restrict__ src, const int* __restrict__ dst,
    const float* __restrict__ xl, const float* __restrict__ xr,
    const float* __restrict__ att, const unsigned* __restrict__ m_enc,
    float* __restrict__ denom, int E)
{
    int e = blockIdx.x * 256 + threadIdx.x;
    if (e >= E) return;
    int s = src[e], d = dst[e];
    float xls[32], lg[8];
    compute_logits(xl, xr, s, d, att, xls, lg);
#pragma unroll
    for (int h = 0; h < HEADS; ++h) {
        float m = dec_f32(m_enc[d * HEADS + h]);
        atomicAdd(&denom[d * HEADS + h], __expf(lg[h] - m));
    }
}

__global__ void __launch_bounds__(256) edge_msg_kernel(
    const int* __restrict__ src, const int* __restrict__ dst,
    const float* __restrict__ xl, const float* __restrict__ xr,
    const float* __restrict__ att, const unsigned* __restrict__ m_enc,
    const float* __restrict__ denom, float* __restrict__ agg, int E)
{
    int e = blockIdx.x * 256 + threadIdx.x;
    if (e >= E) return;
    int s = src[e], d = dst[e];
    float xls[32], lg[8];
    compute_logits(xl, xr, s, d, att, xls, lg);
#pragma unroll
    for (int h = 0; h < HEADS; ++h) {
        float m  = dec_f32(m_enc[d * HEADS + h]);
        float w  = __expf(lg[h] - m);
        float al = w / fmaxf(denom[d * HEADS + h], 1e-16f);
#pragma unroll
        for (int c = 0; c < OUT_C; ++c)
            atomicAdd(&agg[d * HC + h * OUT_C + c], al * xls[h * OUT_C + c]);
    }
}

// ---------------------------------------------------------------------------
// Residual combine + per-layer bias:  h += agg + conv_bias
// ---------------------------------------------------------------------------
__global__ void __launch_bounds__(256) combine_kernel(
    float* __restrict__ h, const float* __restrict__ agg,
    const float* __restrict__ cbias)
{
    int i = blockIdx.x * 256 + threadIdx.x;
    if (i < N_NODES * HC) h[i] += agg[i] + cbias[i & (HC - 1)];
}

// ---------------------------------------------------------------------------
// Final prediction:  out = h @ w_pred + b_pred   (32 x 7, trivial)
// ---------------------------------------------------------------------------
__global__ void __launch_bounds__(256) pred_kernel(
    const float* __restrict__ h, const float* __restrict__ wp,
    const float* __restrict__ bp, float* __restrict__ out)
{
    int n = blockIdx.x * 256 + threadIdx.x;
    if (n >= N_NODES) return;
    float hv[32];
    const float4* hp = (const float4*)(h + (long)n * DIM);
#pragma unroll
    for (int q = 0; q < 8; ++q) {
        float4 t = hp[q];
        hv[4*q] = t.x; hv[4*q+1] = t.y; hv[4*q+2] = t.z; hv[4*q+3] = t.w;
    }
#pragma unroll
    for (int c = 0; c < NUM_CLASS; ++c) {
        float acc = bp[c];
#pragma unroll
        for (int k = 0; k < DIM; ++k) acc += hv[k] * wp[k * NUM_CLASS + c];
        out[n * NUM_CLASS + c] = acc;
    }
}

__global__ void __launch_bounds__(256) fill_u32_kernel(unsigned* p, unsigned v, int n)
{
    int i = blockIdx.x * 256 + threadIdx.x;
    if (i < n) p[i] = v;
}

// ---------------------------------------------------------------------------
extern "C" void kernel_launch(void* const* d_in, const int* in_sizes, int n_in,
                              void* d_out, int out_size, void* d_ws, size_t ws_size,
                              hipStream_t stream)
{
    (void)n_in; (void)out_size; (void)ws_size;

    const float* x      = (const float*)d_in[0];
    const int*   ei     = (const int*)  d_in[1];
    const float* w_proj = (const float*)d_in[2];
    const float* b_proj = (const float*)d_in[3];
    const float* w_l    = (const float*)d_in[4];
    const float* b_l    = (const float*)d_in[5];
    const float* w_r    = (const float*)d_in[6];
    const float* b_r    = (const float*)d_in[7];
    const float* att    = (const float*)d_in[8];
    const float* cbias  = (const float*)d_in[9];
    const float* w_pred = (const float*)d_in[10];
    const float* b_pred = (const float*)d_in[11];

    const int E = in_sizes[1] / 2;
    const int* src = ei;
    const int* dst = ei + E;

    // workspace carve-out (~58 MB)
    float*    h     = (float*)d_ws;
    float*    xl    = h   + (size_t)N_NODES * HC;
    float*    xr    = xl  + (size_t)N_NODES * HC;
    float*    agg   = xr  + (size_t)N_NODES * HC;
    unsigned* m_enc = (unsigned*)(agg + (size_t)N_NODES * HC);
    float*    denom = (float*)(m_enc + (size_t)N_NODES * HEADS);

    const dim3 blk(256);
    const int gRows  = (N_NODES + ROWS_BLK - 1) / ROWS_BLK;
    const int gEdges = (E + 255) / 256;
    const int gNH    = (N_NODES * HEADS + 255) / 256;
    const int gNC    = (N_NODES * HC + 255) / 256;

    proj_gemm_kernel<<<gRows, blk, 0, stream>>>(x, w_proj, b_proj, h);

    for (int L = 0; L < 2; ++L) {
        xlxr_kernel<<<gRows, blk, 0, stream>>>(
            h, w_l + (size_t)L * DIM * HC, b_l + (size_t)L * HC,
               w_r + (size_t)L * DIM * HC, b_r + (size_t)L * HC, xl, xr);

        fill_u32_kernel<<<gNH, blk, 0, stream>>>(m_enc, 0x007FFFFFu, N_NODES * HEADS); // enc(-inf)
        fill_u32_kernel<<<gNH, blk, 0, stream>>>((unsigned*)denom, 0u, N_NODES * HEADS);
        fill_u32_kernel<<<gNC, blk, 0, stream>>>((unsigned*)agg, 0u, N_NODES * HC);

        const float* attL = att + (size_t)L * HEADS * OUT_C;
        edge_max_kernel<<<gEdges, blk, 0, stream>>>(src, dst, xl, xr, attL, m_enc, E);
        edge_sum_kernel<<<gEdges, blk, 0, stream>>>(src, dst, xl, xr, attL, m_enc, denom, E);
        edge_msg_kernel<<<gEdges, blk, 0, stream>>>(src, dst, xl, xr, attL, m_enc, denom, agg, E);

        combine_kernel<<<gNC, blk, 0, stream>>>(h, agg, cbias + (size_t)L * HC);
    }

    pred_kernel<<<(N_NODES + 255) / 256, blk, 0, stream>>>(h, w_pred, b_pred, (float*)d_out);
}
